// StudentQ_4337916970101
// MI455X (gfx1250) — matrix operand order repaired and checked by reading, not compiled, not run on hardware
//
#include <hip/hip_runtime.h>

// GRU (B=4096,T=512,H=128) + MLP head for MI455X (gfx1250, wave32, WMMA).
// Compute/latency-bound recurrence (~206 GFLOP vs ~50MB HBM). v3:
//  - wave w owns gate columns [16w,16w+16) of r,z,n (N-tiles {w,w+8,w+16});
//    gate nonlinearity entirely in registers on the WMMA accumulators.
//  - W_hh and zero-padded w_ih stationary in VGPRs as bf16 B-fragments.
//  - h state per-lane in v8f; cross-wave h exchange via DOUBLE-BUFFERED bf16
//    A-fragment LDS buffers -> ONE barrier per step.
//  - sigmoid/tanh on the fast v_exp/v_rcp trans pipe (co-executes with WMMA).

#define B_TOT   4096
#define T_TOT   512
#define H_DIM   128
#define HID_DIM 256
#define N_ACT   12
#define BTILE   16
#define NTHREADS 256

typedef __attribute__((ext_vector_type(16))) __bf16 v16bf;
typedef __attribute__((ext_vector_type(8)))  float  v8f;

union BF16x16 { v16bf v; __bf16 b[16]; unsigned short u[16]; };

__device__ __forceinline__ float sigf(float x) {
    // v_exp_f32 + v_rcp_f32 fast path (no IEEE divide sequence)
    return __builtin_amdgcn_rcpf(1.0f + __expf(-x));
}

__launch_bounds__(NTHREADS)
__global__ void gru_head_kernel(const float* __restrict__ x,        // [B,T,6]
                                const float* __restrict__ last_obs, // [B,3]
                                const int*   __restrict__ lengths,  // [B]
                                const float* __restrict__ w_ih,     // [3H,6]
                                const float* __restrict__ w_hh,     // [3H,H]
                                const float* __restrict__ b_ih,     // [3H]
                                const float* __restrict__ b_hh,     // [3H]
                                const float* __restrict__ w1,       // [256,131]
                                const float* __restrict__ b1,       // [256]
                                const float* __restrict__ w2,       // [12,256]
                                const float* __restrict__ b2,       // [12]
                                float*       __restrict__ out)      // [B,12]
{
    __shared__ __align__(32) __bf16 s_hfrag[2][4][32][16];   // double-buffered A-frags (8KB)
    __shared__ __align__(32) float  s_hF[BTILE][H_DIM];      // final h (8KB)
    __shared__ float s_hid[BTILE * HID_DIM];                 // MLP hidden (16KB)
    __shared__ float s_xt[2][BTILE * 6];
    __shared__ float s_lobs[BTILE * 3];

    const int tid  = threadIdx.x;
    const int lane = tid & 31;
    const int wave = tid >> 5;                 // 0..7
    const int b0   = blockIdx.x * BTILE;

    // ---- LDS init + first x slice ----
    if (tid < BTILE * 3) s_lobs[tid] = last_obs[(size_t)b0 * 3 + tid];
    for (int i = tid; i < 2 * 4 * 32 * 16; i += NTHREADS)
        ((unsigned short*)s_hfrag)[i] = 0;
    if (tid < BTILE * 6)
        s_xt[0][tid] = x[((size_t)(b0 + tid / 6) * T_TOT) * 6 + (tid % 6)];

    // ---- per-lane constants ----
    const int colB  = lane & 15;
    const int c     = wave * 16 + colB;        // gate-local column 0..127
    const int mBase = (lane >> 4) * 8;         // C-layout row base
    const float br  = b_ih[c]       + b_hh[c];
    const float bz  = b_ih[c + 128] + b_hh[c + 128];
    const float bin = b_ih[c + 256];
    const float bhn = b_hh[c + 256];
    int lenv[8];
    #pragma unroll
    for (int v = 0; v < 8; v++) lenv[v] = lengths[b0 + mBase + v];

    // ---- stationary weights in VGPRs ----
    // B[k][n] = w_hh[n][k]; gate g -> N-tile g*8+wave.
    // B-frag: lane holds n = nt*16+(lane&15), K = (lane>>4)*16 + e.
    BF16x16 wf[12];
    #pragma unroll
    for (int g = 0; g < 3; g++) {
        const int n = (g * 8 + wave) * 16 + colB;
        #pragma unroll
        for (int kt = 0; kt < 4; kt++) {
            const float* wp = &w_hh[(size_t)n * H_DIM + kt * 32 + (lane >> 4) * 16];
            #pragma unroll
            for (int e = 0; e < 16; e++)
                wf[g * 4 + kt].b[e] = (__bf16)wp[e];
        }
    }
    // x-side B-frags (K=6 zero-padded to 32): nonzero only lanes 0-15, e<6.
    BF16x16 wfX[3];
    #pragma unroll
    for (int g = 0; g < 3; g++) {
        const int n = (g * 8 + wave) * 16 + colB;
        #pragma unroll
        for (int e = 0; e < 16; e++) {
            if (lane < 16 && e < 6) wfX[g].b[e] = (__bf16)w_ih[(size_t)n * 6 + e];
            else                    wfX[g].u[e] = 0;
        }
    }

    // h-frag write slot: col c fixed per lane; only row (=mBase+v) varies.
    const int kk = c & 31, ktc = c >> 5;
    const int halfc = kk >> 4;
    const int ec = kk & 15;
    __bf16* hfW[2] = { &s_hfrag[0][ktc][mBase + 16 * halfc][ec],
                       &s_hfrag[1][ktc][mBase + 16 * halfc][ec] };

    v8f hstate = {};
    __syncthreads();

    // ================= time loop (one barrier per step) =================
    for (int t = 0; t < T_TOT; t++) {
        const int cur = t & 1, nxt = cur ^ 1;

        // A-frags of h (buffer `cur`) + x A-frag (rows 0-15, K=0..5)
        BF16x16 afr[4];
        #pragma unroll
        for (int kt = 0; kt < 4; kt++)
            afr[kt].v = *(const v16bf*)&s_hfrag[cur][kt][lane][0];
        BF16x16 afx;
        #pragma unroll
        for (int e = 0; e < 16; e++) afx.u[e] = 0;
        if (lane < 16) {
            #pragma unroll
            for (int e = 0; e < 6; e++) afx.b[e] = (__bf16)s_xt[cur][lane * 6 + e];
        }

        v8f accR = {}, accZ = {}, accNH = {}, accNX = {};
        accR  = __builtin_amdgcn_wmma_f32_16x16x32_bf16(false, afx.v, false, wfX[0].v, (short)0, accR,  false, false);
        accZ  = __builtin_amdgcn_wmma_f32_16x16x32_bf16(false, afx.v, false, wfX[1].v, (short)0, accZ,  false, false);
        accNX = __builtin_amdgcn_wmma_f32_16x16x32_bf16(false, afx.v, false, wfX[2].v, (short)0, accNX, false, false);
        #pragma unroll
        for (int kt = 0; kt < 4; kt++) {
            accR  = __builtin_amdgcn_wmma_f32_16x16x32_bf16(false, afr[kt].v, false, wf[0 + kt].v, (short)0, accR,  false, false);
            accZ  = __builtin_amdgcn_wmma_f32_16x16x32_bf16(false, afr[kt].v, false, wf[4 + kt].v, (short)0, accZ,  false, false);
            accNH = __builtin_amdgcn_wmma_f32_16x16x32_bf16(false, afr[kt].v, false, wf[8 + kt].v, (short)0, accNH, false, false);
        }

        // stage x for t+1 into `nxt` (overlaps with gate math) + prefetch t+2
        if (t + 1 < T_TOT && tid < BTILE * 6) {
            const size_t base = ((size_t)(b0 + tid / 6) * T_TOT + (t + 1)) * 6 + (tid % 6);
            s_xt[nxt][tid] = x[base];
            if (t + 2 < T_TOT) __builtin_prefetch(&x[base + 6], 0, 1);
        }

        // ---- gates entirely in registers ----
        __bf16* wp = hfW[nxt];
        #pragma unroll
        for (int v = 0; v < 8; v++) {
            const float r   = sigf(accR[v] + br);
            const float z   = sigf(accZ[v] + bz);
            const float y   = accNX[v] + bin + r * (accNH[v] + bhn);
            const float nn  = 2.0f * sigf(2.0f * y) - 1.0f;   // tanh
            const float ho  = hstate[v];
            const float upd = nn + z * (ho - nn);             // (1-z)n + z*h
            const float hw  = (t < lenv[v]) ? upd : ho;
            hstate[v] = hw;
            wp[v * 16] = (__bf16)hw;                          // +32B per row
        }

        __syncthreads();   // publish hfrag[nxt]/s_xt[nxt]; protect hfrag[cur]
    }

    // ---- publish final h ----
    #pragma unroll
    for (int v = 0; v < 8; v++) s_hF[mBase + v][c] = hstate[v];
    __syncthreads();

    // ================= MLP head =================
    {
        float acc16[16];
        const float bj = b1[tid];
        #pragma unroll
        for (int r = 0; r < 16; r++) acc16[r] = bj;
        const float* w1r = &w1[(size_t)tid * 131];
        for (int k = 0; k < H_DIM; k++) {
            const float wv = w1r[k];
            #pragma unroll
            for (int r = 0; r < 16; r++) acc16[r] += s_hF[r][k] * wv;
        }
        #pragma unroll
        for (int k = 0; k < 3; k++) {
            const float wv = w1r[128 + k];
            #pragma unroll
            for (int r = 0; r < 16; r++) acc16[r] += s_lobs[r * 3 + k] * wv;
        }
        #pragma unroll
        for (int r = 0; r < 16; r++) s_hid[r * HID_DIM + tid] = fmaxf(acc16[r], 0.0f);
    }
    __syncthreads();

    if (tid < BTILE * N_ACT) {
        const int r = tid / N_ACT, a = tid % N_ACT;
        float s = b2[a];
        const float* w2r = &w2[(size_t)a * HID_DIM];
        for (int k = 0; k < HID_DIM; k++) s += s_hid[r * HID_DIM + k] * w2r[k];
        out[(size_t)(b0 + r) * N_ACT + a] = s;
    }
}

extern "C" void kernel_launch(void* const* d_in, const int* in_sizes, int n_in,
                              void* d_out, int out_size, void* d_ws, size_t ws_size,
                              hipStream_t stream) {
    const float* x        = (const float*)d_in[0];
    const float* last_obs = (const float*)d_in[1];
    const int*   lengths  = (const int*)  d_in[2];
    const float* w_ih     = (const float*)d_in[3];
    const float* w_hh     = (const float*)d_in[4];
    const float* b_ih     = (const float*)d_in[5];
    const float* b_hh     = (const float*)d_in[6];
    const float* w1       = (const float*)d_in[7];
    const float* b1       = (const float*)d_in[8];
    const float* w2       = (const float*)d_in[9];
    const float* b2       = (const float*)d_in[10];
    float* out = (float*)d_out;

    dim3 grid(B_TOT / BTILE);   // 256 blocks, 16 batch rows each
    dim3 block(NTHREADS);       // 8 waves (wave32)
    gru_head_kernel<<<grid, block, 0, stream>>>(x, last_obs, lengths,
                                                w_ih, w_hh, b_ih, b_hh,
                                                w1, b1, w2, b2, out);
}
